// ROIHead_893353198234
// MI455X (gfx1250) — matrix operand order repaired and checked by reading, not compiled, not run on hardware
//
#include <hip/hip_runtime.h>

// ---------------- problem constants ----------------
#define FEAT_HW   38
#define NPROP     256
#define CHANS     256
#define PP        7
#define KWIN      8
#define DIN       12544      // 256*7*7
#define DKSTEPS   392        // 12544/32
#define HID       1024
#define NCLS      21
#define DWH_CLAMP 4.1351665565f   // log(1000/16)

typedef __attribute__((ext_vector_type(16))) __bf16 v16bf;
typedef __attribute__((ext_vector_type(8)))  __bf16 v8bf;
typedef __attribute__((ext_vector_type(8)))  float  v8f;
typedef __attribute__((ext_vector_type(4)))  float  v4f;
typedef __attribute__((ext_vector_type(8)))  unsigned int v8u;

// Swizzled activation layout ("fragment-linear"):
//   element (row m, k) lives at
//     frag = (m>>4)*ksteps + (k>>5)          (one 16x32 WMMA A-fragment)
//     lane = (m&15) + ((k&8) ? 16 : 0)
//     elem = (k&7)  + ((k&16) ? 8 : 0)
//     addr = frag*512 + lane*16 + elem       (in bf16 elements)
// so in the GEMM, lane L loads its 16 bf16 (32B) contiguously at frag*1KB + L*32.
__device__ __forceinline__ size_t swz_idx(int m, int k, int ksteps) {
  const int lane = (m & 15) + ((k & 8) ? 16 : 0);
  const int elem = (k & 7) + ((k & 16) ? 8 : 0);
  return ((size_t)(m >> 4) * ksteps + (k >> 5)) * 512 + lane * 16 + elem;
}

// Pack 16 floats -> v16bf using packed RNE converts when available.
__device__ __forceinline__ v16bf pack16_bf16(const float* wf) {
#if __has_builtin(__builtin_amdgcn_cvt_pk_bf16_f32)
  v8u u;
#pragma unroll
  for (int i = 0; i < 8; ++i)
    u[i] = __builtin_bit_cast(unsigned int,
            __builtin_amdgcn_cvt_pk_bf16_f32(wf[2 * i], wf[2 * i + 1]));
  return __builtin_bit_cast(v16bf, u);
#else
  v16bf b;
#pragma unroll
  for (int i = 0; i < 16; ++i) b[i] = (__bf16)wf[i];
  return b;
#endif
}

// ---------------------------------------------------------------------------
// Kernel 1: quantized max ROI pooling (torchvision roi_pool semantics),
// writing bf16 activations directly in the WMMA-fragment swizzled layout.
// One block per proposal, one thread per channel. Feature map is L2-resident.
// ---------------------------------------------------------------------------
__global__ __launch_bounds__(256) void roi_pool_kernel(
    const float* __restrict__ feat, const float* __restrict__ prop,
    const int* __restrict__ image_h, __bf16* __restrict__ A0)
{
  const int n = blockIdx.x;
  const int c = threadIdx.x;
  const float ih = (float)image_h[0];
  const float scale = exp2f(rintf(log2f((float)FEAT_HW / ih)));   // 1/16

  const float p0 = prop[n * 4 + 0], p1 = prop[n * 4 + 1];
  const float p2 = prop[n * 4 + 2], p3 = prop[n * 4 + 3];
  const int x1 = (int)rintf(p0 * scale);
  const int y1 = (int)rintf(p1 * scale);
  const int x2 = (int)rintf(p2 * scale);
  const int y2 = (int)rintf(p3 * scale);
  const float roi_w = (float)max(x2 - x1 + 1, 1);
  const float roi_h = (float)max(y2 - y1 + 1, 1);
  const float bsh = roi_h / (float)PP;
  const float bsw = roi_w / (float)PP;

  const float* f = feat + (size_t)c * FEAT_HW * FEAT_HW;

  for (int py = 0; py < PP; ++py) {
    const int hs = min(max((int)floorf((float)py * bsh) + y1, 0), FEAT_HW);
    const int he = min(max((int)ceilf((float)(py + 1) * bsh) + y1, 0), FEAT_HW);
    for (int px = 0; px < PP; ++px) {
      const int wsx = min(max((int)floorf((float)px * bsw) + x1, 0), FEAT_HW);
      const int we  = min(max((int)ceilf((float)(px + 1) * bsw) + x1, 0), FEAT_HW);
      float m = -INFINITY;
      for (int kh = 0; kh < KWIN; ++kh) {
        const int h = hs + kh;
        if (h >= he) break;
        const float* fr = f + h * FEAT_HW;
        for (int kw = 0; kw < KWIN; ++kw) {
          const int w = wsx + kw;
          if (w >= we) break;
          m = fmaxf(m, fr[w]);
        }
      }
      if (m == -INFINITY) m = 0.0f;
      const int kidx = c * (PP * PP) + py * PP + px;
      A0[swz_idx(n, kidx, DKSTEPS)] = (__bf16)m;
    }
  }
}

// ---------------------------------------------------------------------------
// Kernel 2: WMMA GEMM  Out[256][J] = act( A[256][K](bf16,swizzled) @ W[J][K]^T + b )
//
// Block = 8 waves = two 16-column j-tiles (4 waves each). Wave owns 4 M-tiles
// -> 4 accumulators, 4 v_wmma_f32_16x16x32_bf16 per B fragment. W strips are
// disjoint per j-tile, so each W element is streamed once per layer. No B
// masking needed: garbage columns (lj>=J) only affect unstored outputs.
// Software-pipelined with a peeled final step: the steady-state loop loads
// step ks+1 unconditionally (pure induction-variable addressing), then runs
// the WMMAs of step ks on already-resident registers.
// ---------------------------------------------------------------------------
struct Frags { v16bf a0, a1, a2, a3, b; };

__device__ __forceinline__ Frags load_frags(
    const float* __restrict__ wrow,
    const __bf16* __restrict__ af0, const __bf16* __restrict__ af1,
    const __bf16* __restrict__ af2, const __bf16* __restrict__ af3, int ks)
{
  Frags f;
  // A fragments: one 32B contiguous run per lane, per M-tile (2x b128).
  const v8bf* p0 = (const v8bf*)(af0 + (size_t)ks * 512);
  const v8bf* p1 = (const v8bf*)(af1 + (size_t)ks * 512);
  const v8bf* p2 = (const v8bf*)(af2 + (size_t)ks * 512);
  const v8bf* p3 = (const v8bf*)(af3 + (size_t)ks * 512);
  v8bf l0 = p0[0], h0 = p0[1];
  v8bf l1 = p1[0], h1 = p1[1];
  v8bf l2 = p2[0], h2 = p2[1];
  v8bf l3 = p3[0], h3 = p3[1];
  // B fragment: fp32 of the W row at k offsets 0..7 and 16..23 (4x b128) -> packed bf16.
  const v4f* wp = (const v4f*)(wrow + ks * 32);
  v4f q0 = wp[0], q1 = wp[1], q2 = wp[4], q3 = wp[5];
  float wf[16];
#pragma unroll
  for (int i = 0; i < 4; ++i) {
    wf[i] = q0[i]; wf[4 + i] = q1[i]; wf[8 + i] = q2[i]; wf[12 + i] = q3[i];
  }
  f.a0 = __builtin_shufflevector(l0, h0, 0,1,2,3,4,5,6,7,8,9,10,11,12,13,14,15);
  f.a1 = __builtin_shufflevector(l1, h1, 0,1,2,3,4,5,6,7,8,9,10,11,12,13,14,15);
  f.a2 = __builtin_shufflevector(l2, h2, 0,1,2,3,4,5,6,7,8,9,10,11,12,13,14,15);
  f.a3 = __builtin_shufflevector(l3, h3, 0,1,2,3,4,5,6,7,8,9,10,11,12,13,14,15);
  f.b  = pack16_bf16(wf);
  return f;
}

__device__ __forceinline__ void wmma4(const Frags& f, v8f& acc0, v8f& acc1,
                                      v8f& acc2, v8f& acc3)
{
  acc0 = __builtin_amdgcn_wmma_f32_16x16x32_bf16(false, f.a0, false, f.b,
                                                 (short)0, acc0, false, false);
  acc1 = __builtin_amdgcn_wmma_f32_16x16x32_bf16(false, f.a1, false, f.b,
                                                 (short)0, acc1, false, false);
  acc2 = __builtin_amdgcn_wmma_f32_16x16x32_bf16(false, f.a2, false, f.b,
                                                 (short)0, acc2, false, false);
  acc3 = __builtin_amdgcn_wmma_f32_16x16x32_bf16(false, f.a3, false, f.b,
                                                 (short)0, acc3, false, false);
}

__global__ __launch_bounds__(256) void wmma_gemm_kernel(
    const __bf16* __restrict__ A, const float* __restrict__ W,
    const float* __restrict__ bias, void* __restrict__ Out,
    int K, int J, int do_relu, int out_swz)
{
  const int lane   = threadIdx.x & 31;
  const int wave   = threadIdx.x >> 5;
  const int jt     = blockIdx.x * 2 + (wave >> 2);
  const int wsub   = wave & 3;
  const int ksteps = K >> 5;
  const int lj     = jt * 16 + (lane & 15);
  const bool jvalid = (lj < J);
  const int j      = jvalid ? lj : (J - 1);   // clamp: loads stay in bounds
  const int koffB  = (lane < 16) ? 0 : 8;
  const int mt0    = wsub * 4;

  const float* wrow = W + (size_t)j * K + koffB;
  const __bf16* af0 = A + (size_t)(mt0 + 0) * ksteps * 512 + lane * 16;
  const __bf16* af1 = A + (size_t)(mt0 + 1) * ksteps * 512 + lane * 16;
  const __bf16* af2 = A + (size_t)(mt0 + 2) * ksteps * 512 + lane * 16;
  const __bf16* af3 = A + (size_t)(mt0 + 3) * ksteps * 512 + lane * 16;

  v8f acc0 = {}, acc1 = {}, acc2 = {}, acc3 = {};

  Frags cur = load_frags(wrow, af0, af1, af2, af3, 0);
#pragma unroll 2
  for (int ks = 0; ks < ksteps - 1; ++ks) {
    // stream-prefetch W eight k-steps ahead (global_prefetch_b8)
    __builtin_prefetch(wrow + (ks + 8) * 32, 0, 1);
    // issue next step's loads before this step's matrix ops (latency hiding)
    Frags nxt = load_frags(wrow, af0, af1, af2, af3, ks + 1);
    wmma4(cur, acc0, acc1, acc2, acc3);
    cur = nxt;
  }
  wmma4(cur, acc0, acc1, acc2, acc3);   // peeled final step: no loads

  // ---- epilogue: bias, ReLU, store per C/D layout ------------------------
  const float bj = jvalid ? bias[j] : 0.0f;
  const int rbase = (lane < 16) ? 0 : 8;
  const int jkst = J >> 5;

  auto store_tile = [&](const v8f& acc, int mt) {
#pragma unroll
    for (int r = 0; r < 8; ++r) {
      const int row = mt * 16 + r + rbase;
      float v = acc[r] + bj;
      if (do_relu) v = fmaxf(v, 0.0f);
      if (jvalid) {
        if (out_swz) {
          ((__bf16*)Out)[swz_idx(row, lj, jkst)] = (__bf16)v;
        } else {
          ((float*)Out)[(size_t)row * J + lj] = v;
        }
      }
    }
  };
  store_tile(acc0, mt0 + 0);
  store_tile(acc1, mt0 + 1);
  store_tile(acc2, mt0 + 2);
  store_tile(acc3, mt0 + 3);
}

// ---------------------------------------------------------------------------
// Kernel 3: softmax over 21 classes + box decoding + output packing.
// d_out = boxes[256*20*4] || scores[256*20] || labels[256*20]  (floats)
// ---------------------------------------------------------------------------
__global__ __launch_bounds__(256) void postprocess_kernel(
    const float* __restrict__ cls, const float* __restrict__ box,
    const float* __restrict__ prop, const int* __restrict__ ihp,
    const int* __restrict__ iwp, float* __restrict__ out)
{
  const int n = threadIdx.x;           // 256 proposals, one thread each
  const float ih = (float)ihp[0];
  const float iw = (float)iwp[0];

  float sc[NCLS];
  float mx = -INFINITY;
#pragma unroll
  for (int c = 0; c < NCLS; ++c) { sc[c] = cls[n * NCLS + c]; mx = fmaxf(mx, sc[c]); }
  float s = 0.0f;
#pragma unroll
  for (int c = 0; c < NCLS; ++c) { sc[c] = expf(sc[c] - mx); s += sc[c]; }
  const float inv = 1.0f / s;

  const float px1 = prop[n * 4 + 0], py1 = prop[n * 4 + 1];
  const float px2 = prop[n * 4 + 2], py2 = prop[n * 4 + 3];
  const float pw = px2 - px1, ph = py2 - py1;
  const float cx = px1 + 0.5f * pw, cy = py1 + 0.5f * ph;

  for (int cl = 1; cl < NCLS; ++cl) {
    const float* bp = box + n * (NCLS * 4) + cl * 4;
    const float dx = bp[0], dy = bp[1];
    const float dw = fminf(bp[2], DWH_CLAMP);
    const float dh = fminf(bp[3], DWH_CLAMP);
    const float pcx = dx * pw + cx;
    const float pcy = dy * ph + cy;
    const float pwid = expf(dw) * pw;
    const float phei = expf(dh) * ph;
    const float bx1 = fminf(fmaxf(pcx - 0.5f * pwid, 0.0f), iw);
    const float by1 = fminf(fmaxf(pcy - 0.5f * phei, 0.0f), ih);
    const float bx2 = fminf(fmaxf(pcx + 0.5f * pwid, 0.0f), iw);
    const float by2 = fminf(fmaxf(pcy + 0.5f * phei, 0.0f), ih);
    const int o = n * (NCLS - 1) + (cl - 1);
    out[o * 4 + 0] = bx1;
    out[o * 4 + 1] = by1;
    out[o * 4 + 2] = bx2;
    out[o * 4 + 3] = by2;
    out[NPROP * (NCLS - 1) * 4 + o] = sc[cl] * inv;   // scores
    out[NPROP * (NCLS - 1) * 5 + o] = (float)cl;      // labels
  }
}

// ---------------------------------------------------------------------------
// Workspace layout (bytes):
//   A0  bf16 swz [256][12544] @ 0          (6,422,528)
//   X1  bf16 swz [256][1024]  @ 6,422,528  (  524,288)
//   X2  bf16 swz [256][1024]  @ 6,946,816  (  524,288)
//   CLS f32      [256][21]    @ 7,471,104  (   21,504)
//   BOX f32      [256][84]    @ 7,492,608  (   86,016)
// ---------------------------------------------------------------------------
extern "C" void kernel_launch(void* const* d_in, const int* in_sizes, int n_in,
                              void* d_out, int out_size, void* d_ws, size_t ws_size,
                              hipStream_t stream) {
  const float* feat = (const float*)d_in[0];
  const float* prop = (const float*)d_in[1];
  const float* w1   = (const float*)d_in[2];
  const float* b1   = (const float*)d_in[3];
  const float* w2   = (const float*)d_in[4];
  const float* b2   = (const float*)d_in[5];
  const float* wc   = (const float*)d_in[6];
  const float* bc   = (const float*)d_in[7];
  const float* wb   = (const float*)d_in[8];
  const float* bb   = (const float*)d_in[9];
  const int*   ih   = (const int*)d_in[10];
  const int*   iw   = (const int*)d_in[11];

  char* ws = (char*)d_ws;
  __bf16* A0  = (__bf16*)(ws);
  __bf16* X1  = (__bf16*)(ws + 6422528);
  __bf16* X2  = (__bf16*)(ws + 6946816);
  float*  CLS = (float*)(ws + 7471104);
  float*  BOX = (float*)(ws + 7492608);

  roi_pool_kernel<<<NPROP, CHANS, 0, stream>>>(feat, prop, ih, A0);

  // FC1: [256,12544] x [1024,12544]^T -> bf16 swz, ReLU  (32 blocks x 2 jt)
  wmma_gemm_kernel<<<32, 256, 0, stream>>>(A0, w1, b1, X1, DIN, HID, 1, 1);
  // FC2: [256,1024] x [1024,1024]^T -> bf16 swz, ReLU
  wmma_gemm_kernel<<<32, 256, 0, stream>>>(X1, w2, b2, X2, HID, HID, 1, 1);
  // cls head: J=21 (2 j-tiles -> 1 block, guarded lanes)
  wmma_gemm_kernel<<<1, 256, 0, stream>>>(X2, wc, bc, CLS, HID, NCLS, 0, 0);
  // box head: J=84 (6 j-tiles -> 3 blocks, guarded lanes)
  wmma_gemm_kernel<<<3, 256, 0, stream>>>(X2, wb, bb, BOX, HID, NCLS * 4, 0, 0);

  postprocess_kernel<<<1, 256, 0, stream>>>(CLS, BOX, prop, ih, iw, (float*)d_out);
}
